// Estimator_4037269258477
// MI455X (gfx1250) — matrix operand order repaired and checked by reading, not compiled, not run on hardware
//
#include <hip/hip_runtime.h>

// ---------------------------------------------------------------------------
// Types / constants
// ---------------------------------------------------------------------------
typedef __attribute__((ext_vector_type(16))) _Float16 v16h;
typedef __attribute__((ext_vector_type(8)))  _Float16 v8h;
typedef __attribute__((ext_vector_type(8)))  float    v8f;

#define BUFSH   20u               // floats per [16,16,64,64] buffer = 1<<20
#define HIDSH   19u               // floats per [16,8,64,64]  buffer = 1<<19
#define NHID    10                // max concurrent passes per merged step (9 used)
#define HEAT_SZ 851968u           // 16*13*4096
#define INTER_SZ 13631488u        // 16*13*16*4096

struct PassBatch {
  int n;
  int widx[NHID];     // global pass index (weights + pass BN)
  int src[NHID];      // source acc buffer (0..25)
  int dsttype[NHID];  // 0 = accumulate into acc[dst]; 1 = inter_a[label dst]; 2 = inter_b
  int dst[NHID];
};

// WMMA 16x16x32 f16 operand K-offsets, contiguous k order shared by A and B (wave32).
// A (16x32, M=lane&15): lane<16 covers K k0+0..15, lane>=16 covers K k0+16..31.
__device__ __forceinline__ int a_koff(int lane, int j) {
  return j + ((lane >> 4) << 4);
}
// B (32x16, N=lane&15): lane<16 covers K k0+0..15, lane>=16 covers K k0+16..31.
__device__ __forceinline__ int b_koff(int lane, int j) {
  return j + ((lane >> 4) << 4);
}

// ---------------------------------------------------------------------------
// Kernel 1: all 26 pre-projections as one WMMA GEMM.
//   Y[pi][16, 65536] = pre_w[pi][16,256] * X[256, 65536]   (N = b*4096 + pix)
// One wave per 16x16 output tile; K=256 in 8 WMMA steps. B loads are already
// coalesced (16 lanes -> 16 consecutive pixels) and each (k,pix) is read by
// exactly one lane, so no LDS staging is needed here.
// ---------------------------------------------------------------------------
__global__ __launch_bounds__(256) void proj_gemm(
    const float* __restrict__ x, const float* __restrict__ pre_w,
    float* __restrict__ acc) {
  const int lane  = threadIdx.x & 31;
  const int wave  = threadIdx.x >> 5;
  const int ntile = blockIdx.x * 8 + wave;     // 0..4095
  const int pi    = blockIdx.y;                // 0..25
  const int n0    = ntile << 4;
  const int b     = n0 >> 12;                  // 4096 pixels per image
  const int pix   = (n0 & 4095) + (lane & 15);
  const int m     = lane & 15;
  const float* wrow = pre_w + (size_t)(pi * 16 + m) * 256;
  const float* xcol = x + ((size_t)b << 20) + pix;   // batch stride 256*4096
  v8f c = {};
  for (int k0 = 0; k0 < 256; k0 += 32) {
    v16h av, bv;
#pragma unroll
    for (int j = 0; j < 16; ++j) av[j] = (_Float16)wrow[k0 + a_koff(lane, j)];
#pragma unroll
    for (int j = 0; j < 16; ++j)
      bv[j] = (_Float16)xcol[(size_t)(k0 + b_koff(lane, j)) << 12];
    c = __builtin_amdgcn_wmma_f32_16x16x32_f16(false, av, false, bv,
                                               (short)0, c, false, false);
  }
  const int hi = (lane >> 4) << 3;             // D rows: r (lanes<16) / r+8
  float* dst = acc + ((size_t)pi << BUFSH) + ((size_t)b << 16) + pix;
#pragma unroll
  for (int r = 0; r < 8; ++r) dst[(size_t)(r + hi) << 12] = c[r];
}

// ---------------------------------------------------------------------------
// Kernel 2: activation.  acc[buf] = relu(bn(acc[buf] + ahead_msg[br,lab]))
// grid.y = 2 (branch A / branch B of the same step). In place.
// ---------------------------------------------------------------------------
__global__ __launch_bounds__(256) void act_bn_relu(
    float* __restrict__ acc, const float* __restrict__ ahead,
    const float* __restrict__ g, const float* __restrict__ bb,
    const float* __restrict__ mm, const float* __restrict__ vv,
    int buf0, int c0, int buf1, int c1) {
  const int br  = blockIdx.y;
  const int buf = br ? buf1 : buf0;
  const int c   = br ? c1 : c0;
  const int i   = blockIdx.x * 256 + threadIdx.x;   // 0 .. (1<<20)-1 : (b,ch,pix)
  const int ch  = (i >> 12) & 15;
  const int ix  = buf * 16 + ch;                    // pre-BN param index (== pi)
  float val = acc[((size_t)buf << BUFSH) + i] +
              ahead[(((size_t)(br * 13 + c)) << BUFSH) + i];
  const float inv = g[ix] * rsqrtf(vv[ix] + 1e-5f);
  val = val * inv + (bb[ix] - mm[ix] * inv);
  acc[((size_t)buf << BUFSH) + i] = fmaxf(val, 0.f);
}

// ---------------------------------------------------------------------------
// Kernel 3: conv7x7 16->8 + BN + ReLU, implicit-GEMM WMMA with LDS staging.
// Block = 2 output rows x 64 cols of one image. Input patch (8 rows x 72 cols
// x 16ch, zero-padded, f16, channel-innermost) and the full A-operand image
// (exact WMMA register order, taps>=49 and m>=8 pre-zeroed) live in LDS, so
// the hot loop is: 1x b128-pair LDS load (A) + 1x 32B LDS load (B) + WMMA.
// K ordered k = tap*16 + ci  (25 chunks of 32).
// ---------------------------------------------------------------------------
__global__ __launch_bounds__(256) void pass_conv1(
    const float* __restrict__ acc, const float* __restrict__ w1,
    const float* __restrict__ bng, const float* __restrict__ bnb,
    const float* __restrict__ bnm, const float* __restrict__ bnv,
    float* __restrict__ hid, PassBatch pbt) {
  __shared__ __align__(32) _Float16 smA[25 * 32 * 16];  // 25.6 KB
  __shared__ __align__(32) _Float16 smB[8 * 72 * 16];   // 18.4 KB [row][col][ch]
  const int tid    = threadIdx.x;
  const int lane   = tid & 31;
  const int wave   = tid >> 5;
  const int py     = blockIdx.y;
  const int wp     = pbt.widx[py];
  const int b      = blockIdx.x >> 5;          // 32 blocks per image
  const int y0     = (blockIdx.x & 31) << 1;   // 2 rows per block
  const float* src   = acc + ((size_t)pbt.src[py] << BUFSH) + ((size_t)b << 16);
  const float* wbase = w1 + (size_t)wp * (8 * 16 * 49);

  // Stage A operand image: pair p = chunk*32 + lane.
  for (int p = tid; p < 25 * 32; p += 256) {
    const int ck = p >> 5, ln = p & 31, k0 = ck << 5, m = ln & 15;
    alignas(32) _Float16 tmp[16];
#pragma unroll
    for (int j = 0; j < 16; ++j) {
      const int k = k0 + a_koff(ln, j);
      const int tap = k >> 4, ci = k & 15;
      float wv = 0.f;
      if (m < 8 && tap < 49) wv = wbase[(m * 16 + ci) * 49 + tap];
      tmp[j] = (_Float16)wv;
    }
    *(v16h*)&smA[p << 4] = *(const v16h*)tmp;
  }
  // Stage zero-padded input patch: rows y0-3..y0+4, cols -3..68.
  for (int p = tid; p < 8 * 72; p += 256) {
    const int row = p / 72, col = p % 72;
    const int gy = y0 - 3 + row, gx = col - 3;
    const bool ok = ((unsigned)gy < 64u) & ((unsigned)gx < 64u);
    alignas(32) _Float16 tmp[16];
#pragma unroll
    for (int ch = 0; ch < 16; ++ch)
      tmp[ch] = (_Float16)(ok ? src[((size_t)ch << 12) + (gy << 6) + gx] : 0.f);
    *(v16h*)&smB[p << 4] = *(const v16h*)tmp;
  }
  __syncthreads();

  const int ty     = wave >> 2;                       // output row within block
  const int xloc   = ((wave & 3) << 4) + (lane & 15); // output col 0..63
  const int lanehi = lane >> 4;
  v8f c = {};
  for (int chunk = 0; chunk < 25; ++chunk) {
    const int tapb = (chunk << 1) + lanehi;    // lane group covers one whole tap
    const int tapc = tapb < 49 ? tapb : 48;    // A already zero for tap>=49
    const int dy = tapc / 7 - 3, dx = tapc % 7 - 3;
    const int row = ty + 3 + dy;               // in [0,8)
    const int col = xloc + dx + 3;             // in [0,70)
    const v16h av = *(const v16h*)&smA[(chunk * 32 + lane) << 4];
    const v16h bv = *(const v16h*)&smB[(row * 72 + col) << 4];
    c = __builtin_amdgcn_wmma_f32_16x16x32_f16(false, av, false, bv,
                                               (short)0, c, false, false);
  }
  const int hi  = lanehi << 3;
  const int pix = ((y0 + ty) << 6) + xloc;
  float* hdst = hid + ((size_t)py << HIDSH) + ((size_t)b << 15) + pix;
#pragma unroll
  for (int r = 0; r < 8; ++r) {
    const int mo = r + hi;
    if (mo < 8) {
      const int ix = wp * 8 + mo;
      const float inv = bng[ix] * rsqrtf(bnv[ix] + 1e-5f);
      hdst[(size_t)mo << 12] =
          fmaxf(c[r] * inv + (bnb[ix] - bnm[ix] * inv), 0.f);
    }
  }
}

// ---------------------------------------------------------------------------
// Kernel 4: conv7x7 8->16, LDS-staged implicit-GEMM WMMA; accumulates into
// acc[tgt] or writes inter outputs.  K ordered k = tap*8 + ci (13 chunks).
// ---------------------------------------------------------------------------
__global__ __launch_bounds__(256) void pass_conv2(
    float* __restrict__ acc, const float* __restrict__ hid,
    const float* __restrict__ w2, float* __restrict__ dout, PassBatch pbt) {
  __shared__ __align__(32) _Float16 smA[13 * 32 * 16];  // 13.3 KB
  __shared__ __align__(16) _Float16 smB[8 * 72 * 8];    //  9.2 KB [row][col][ch8]
  const int tid    = threadIdx.x;
  const int lane   = tid & 31;
  const int wave   = tid >> 5;
  const int py     = blockIdx.y;
  const int wp     = pbt.widx[py];
  const int b      = blockIdx.x >> 5;
  const int y0     = (blockIdx.x & 31) << 1;
  const float* hsrc  = hid + ((size_t)py << HIDSH) + ((size_t)b << 15);
  const float* wbase = w2 + (size_t)wp * (16 * 8 * 49);

  // Stage A operand image.
  for (int p = tid; p < 13 * 32; p += 256) {
    const int ck = p >> 5, ln = p & 31, k0 = ck << 5, m = ln & 15;
    alignas(32) _Float16 tmp[16];
#pragma unroll
    for (int j = 0; j < 16; ++j) {
      const int k = k0 + a_koff(ln, j);
      const int tap = k >> 3, ci = k & 7;
      tmp[j] = (_Float16)((tap < 49) ? wbase[(m * 8 + ci) * 49 + tap] : 0.f);
    }
    *(v16h*)&smA[p << 4] = *(const v16h*)tmp;
  }
  // Stage zero-padded hidden patch (8 channels).
  for (int p = tid; p < 8 * 72; p += 256) {
    const int row = p / 72, col = p % 72;
    const int gy = y0 - 3 + row, gx = col - 3;
    const bool ok = ((unsigned)gy < 64u) & ((unsigned)gx < 64u);
    alignas(16) _Float16 tmp[8];
#pragma unroll
    for (int ch = 0; ch < 8; ++ch)
      tmp[ch] = (_Float16)(ok ? hsrc[((size_t)ch << 12) + (gy << 6) + gx] : 0.f);
    *(v8h*)&smB[p << 3] = *(const v8h*)tmp;
  }
  __syncthreads();

  const int ty     = wave >> 2;
  const int xloc   = ((wave & 3) << 4) + (lane & 15);
  const int lanehi = lane >> 4;
  v8f c = {};
  for (int chunk = 0; chunk < 13; ++chunk) {
    const v16h av = *(const v16h*)&smA[(chunk * 32 + lane) << 4];
    v16h bv;
#pragma unroll
    for (int half = 0; half < 2; ++half) {     // each 8-elem half = one tap
      const int tap = (chunk << 2) + half + (lanehi << 1);
      const int tapc = tap < 49 ? tap : 48;    // A already zero for tap>=49
      const int dy = tapc / 7 - 3, dx = tapc % 7 - 3;
      const int row = ty + 3 + dy;
      const int col = xloc + dx + 3;
      const v8h hb = *(const v8h*)&smB[(row * 72 + col) << 3];
#pragma unroll
      for (int j = 0; j < 8; ++j) bv[half * 8 + j] = hb[j];
    }
    c = __builtin_amdgcn_wmma_f32_16x16x32_f16(false, av, false, bv,
                                               (short)0, c, false, false);
  }
  const int hi  = lanehi << 3;
  const int pix = ((y0 + ty) << 6) + xloc;
  const int dt  = pbt.dsttype[py];
  const int dv  = pbt.dst[py];
  if (dt == 0) {  // message accumulation (targets distinct within a launch)
    float* dst = acc + ((size_t)dv << BUFSH) + ((size_t)b << 16) + pix;
#pragma unroll
    for (int r = 0; r < 8; ++r) dst[(size_t)(r + hi) << 12] += c[r];
  } else {        // inter output [B,13,16,64,64]
    const size_t off = (dt == 1) ? (size_t)HEAT_SZ : (size_t)HEAT_SZ + INTER_SZ;
    float* dst = dout + off + (((size_t)(b * 13 + dv)) << 16) + pix;
#pragma unroll
    for (int r = 0; r < 8; ++r) dst[(size_t)(r + hi) << 12] = c[r];
  }
}

// ---------------------------------------------------------------------------
// Kernel 5: final heat. heat[b,c,p] = sum_i aft_w[c,i]*relu(bn(concat[res_a,res_b]))
// ---------------------------------------------------------------------------
__constant__ int POSA[13] = {0, 11, 12, 6, 5, 9, 7, 10, 8, 4, 3, 2, 1};
__constant__ int POSB[13] = {12, 0, 1, 6, 7, 2, 4, 3, 5, 8, 9, 10, 11};

__global__ __launch_bounds__(256) void heat_kernel(
    const float* __restrict__ acc, const float* __restrict__ g,
    const float* __restrict__ bb, const float* __restrict__ mm,
    const float* __restrict__ vv, const float* __restrict__ aftw,
    float* __restrict__ dout) {
  const int i   = blockIdx.x * 256 + threadIdx.x;   // 0..851967 : (b, c, pix)
  const int pix = i & 4095;
  const int c   = (i >> 12) % 13;
  const int b   = i / (13 * 4096);
  const int bufA = POSA[c];
  const int bufB = 13 + POSB[c];
  float s = 0.f;
#pragma unroll
  for (int ch = 0; ch < 16; ++ch) {
    const int ix = c * 32 + ch;
    const float r = acc[((size_t)bufA << BUFSH) + (((size_t)(b * 16 + ch)) << 12) + pix];
    const float inv = g[ix] * rsqrtf(vv[ix] + 1e-5f);
    s += aftw[ix] * fmaxf(r * inv + (bb[ix] - mm[ix] * inv), 0.f);
  }
#pragma unroll
  for (int ch = 0; ch < 16; ++ch) {
    const int ix = c * 32 + 16 + ch;
    const float r = acc[((size_t)bufB << BUFSH) + (((size_t)(b * 16 + ch)) << 12) + pix];
    const float inv = g[ix] * rsqrtf(vv[ix] + 1e-5f);
    s += aftw[ix] * fmaxf(r * inv + (bb[ix] - mm[ix] * inv), 0.f);
  }
  dout[i] = s;
}

// ---------------------------------------------------------------------------
// Host-side schedule
// ---------------------------------------------------------------------------
extern "C" void kernel_launch(void* const* d_in, const int* in_sizes, int n_in,
                              void* d_out, int out_size, void* d_ws, size_t ws_size,
                              hipStream_t stream) {
  (void)in_sizes; (void)n_in; (void)out_size; (void)ws_size;
  const float* x      = (const float*)d_in[0];
  const float* ahead  = (const float*)d_in[1];
  const float* pre_w  = (const float*)d_in[2];
  const float* pbg    = (const float*)d_in[3];
  const float* pbb    = (const float*)d_in[4];
  const float* pbm    = (const float*)d_in[5];
  const float* pbv    = (const float*)d_in[6];
  const float* abg    = (const float*)d_in[7];
  const float* abb    = (const float*)d_in[8];
  const float* abm    = (const float*)d_in[9];
  const float* abv    = (const float*)d_in[10];
  const float* aftw   = (const float*)d_in[11];
  const float* w1     = (const float*)d_in[12];
  const float* w2     = (const float*)d_in[13];
  const float* sbg    = (const float*)d_in[14];
  const float* sbb    = (const float*)d_in[15];
  const float* sbm    = (const float*)d_in[16];
  const float* sbv    = (const float*)d_in[17];
  float* out = (float*)d_out;

  float* acc = (float*)d_ws;                     // 26 x (1<<20) floats
  float* hid = acc + ((size_t)26 << BUFSH);      // 10 x (1<<19) floats

  // All 26 projections (message accumulators are seeded with projection output).
  proj_gemm<<<dim3(512, 26), 256, 0, stream>>>(x, pre_w, acc);

  static const int ordA[13] = {1, 13, 12, 11, 10, 5, 4, 7, 9, 6, 8, 2, 3};
  static const int ordB[13] = {2, 3, 6, 8, 7, 9, 4, 5, 10, 11, 12, 13, 1};
  static const int relA[13][5] = {{2,3,7,9,13},{},{},{7,9},{4},{2},{6},{3},{8},{5},{10},{11},{12}};
  static const int relAcnt[13] = {5,0,0,2,1,1,1,1,1,1,1,1,1};
  static const int relB[13][5] = {{},{1,6},{1,8},{5},{10},{7},{1,4},{9},{1,4},{11},{12},{13},{1}};
  static const int relBcnt[13] = {0,2,2,1,1,1,2,1,2,1,1,1,1};
  static const int posA[13] = {0, 11, 12, 6, 5, 9, 7, 10, 8, 4, 3, 2, 1};
  static const int posB[13] = {12, 0, 1, 6, 7, 2, 4, 3, 5, 8, 9, 10, 11};

  int cntA = 0, cntB = 16;                       // branch B continues after A's 16
  for (int i = 0; i < 13; ++i) {
    const int labA = ordA[i], labB = ordB[i];
    // act = relu(bn(proj + accumulated msgs + ahead_msg)), in place, both branches.
    act_bn_relu<<<dim3(4096, 2), 256, 0, stream>>>(
        acc, ahead, pbg, pbb, pbm, pbv, i, labA - 1, 13 + i, labB - 1);

    PassBatch pbt{};
    int n = 0;
    for (int t = 0; t < relAcnt[labA - 1]; ++t) {
      const int tgt = relA[labA - 1][t];
      pbt.widx[n] = cntA++; pbt.src[n] = i;
      pbt.dsttype[n] = 0;   pbt.dst[n] = posA[tgt - 1]; ++n;
    }
    pbt.widx[n] = 32 + i; pbt.src[n] = i;        // inter_a pass
    pbt.dsttype[n] = 1;   pbt.dst[n] = labA - 1; ++n;
    for (int t = 0; t < relBcnt[labB - 1]; ++t) {
      const int tgt = relB[labB - 1][t];
      pbt.widx[n] = cntB++; pbt.src[n] = 13 + i;
      pbt.dsttype[n] = 0;   pbt.dst[n] = 13 + posB[tgt - 1]; ++n;
    }
    pbt.widx[n] = 45 + i; pbt.src[n] = 13 + i;   // inter_b pass
    pbt.dsttype[n] = 2;   pbt.dst[n] = labB - 1; ++n;
    pbt.n = n;

    pass_conv1<<<dim3(512, n), 256, 0, stream>>>(acc, w1, sbg, sbb, sbm, sbv, hid, pbt);
    pass_conv2<<<dim3(512, n), 256, 0, stream>>>(acc, hid, w2, out, pbt);
  }

  // Final heat maps from the in-place activations.
  heat_kernel<<<3328, 256, 0, stream>>>(acc, abg, abb, abm, abv, aftw, out);
}
